// TriangularSylvesterVAE_EdgeAttentionInception_2731599200753
// MI455X (gfx1250) — compile-verified, matrix operand order checked by reading, not run on hardware
//
#include <hip/hip_runtime.h>

// ---------------------------------------------------------------------------
// TriangularSylvesterVAE_EdgeAttentionInception — full forward pass, gfx1250.
// Big GEMMs (EdgeConv messages, GAT linears, amortization heads) run through
// v_wmma_f32_16x16x32_bf16 (bf16 A/B staged in LDS as packed u32 pairs,
// fp32 accumulate). Scatter/softmax/BN/flows are elementwise+atomic kernels.
// Working set (~60 MB) fits MI455X's 192 MB L2, so the chain is L2-resident.
// ---------------------------------------------------------------------------

#define N_NODES 20000
#define N_EDGES 320000
#define N_SL    (N_EDGES + N_NODES)
#define HEADS   3
#define NEG_SLOPE 0.2f
#define EPS_BN  1e-5f
#define NFLOWS  6

typedef __attribute__((ext_vector_type(16))) __bf16 v16bf;
typedef __attribute__((ext_vector_type(8)))  float  v8f;

static __device__ __forceinline__ unsigned f2bf_u(float f) {
  unsigned u = __builtin_bit_cast(unsigned, f);
  return (u + 0x7FFFu + ((u >> 16) & 1u)) >> 16;  // round-to-nearest-even
}
static __device__ __forceinline__ unsigned pack2bf(float lo, float hi) {
  return f2bf_u(lo) | (f2bf_u(hi) << 16);
}

// ---------------------------------------------------------------------------
// Generic bf16-WMMA GEMM.  out[M x Nout] (at colOff within ldOut) = A@W + b.
// MODE 0: A row r = Asrc[r, 0:K]            (node-feature GEMM)
// MODE 1: A row e = [h[dst[e]], h[src[e]] - h[dst[e]]], K = 2F (EdgeConv msg)
// Block = 4 waves = 64 rows; each wave owns a 16-row strip and NCT col tiles.
// K is consumed in chunks of 32 staged into LDS as u32-packed bf16 pairs.
// A tile is row-major-in-K-pairs; B is staged n-major so its fragment mapping
// is identical to A's.  Fragment VGPR v <- packed index (v<4?v:v+4)+4*half
// (CDNA5 ISA §7.12.2 16-bit A 16x32 layout; C/D per the v8f table).
// ---------------------------------------------------------------------------
template <int NCT, int MODE>
__global__ void __launch_bounds__(128)
gemm_bf16_wmma_kernel(int M, int K, int F, int Nout,
                      const float* __restrict__ A, int ldA,
                      const int* __restrict__ srcI, const int* __restrict__ dstI,
                      const float* __restrict__ W, const float* __restrict__ bias,
                      float* __restrict__ out, int ldOut, int colOff) {
  __shared__ unsigned As[64][17];   // 64 rows x 16 packed-u32 (=32 bf16), pad 1
  __shared__ unsigned Bs[96][17];   // up-to-96 cols x 16 packed-u32, pad 1

  const int tid  = threadIdx.x;
  const int wave = tid >> 5;
  const int lane = tid & 31;
  const int wrow = wave * 16;                  // row strip inside block tile
  const int half = lane >> 4;
  const int mloc = lane & 15;
  const int row0 = blockIdx.x * 64;
  const int NL   = NCT * 16;

  v8f acc[NCT];
#pragma unroll
  for (int t = 0; t < NCT; ++t)
#pragma unroll
    for (int v = 0; v < 8; ++v) acc[t][v] = 0.0f;

  const int Kc = (K + 31) >> 5;
  for (int kc = 0; kc < Kc; ++kc) {
    const int k0 = kc * 32;
    // -- stage A chunk (64 rows x 16 packed pairs) --
    for (int i = tid; i < 64 * 16; i += 128) {
      const int r = i >> 4, jp = i & 15;
      const int e = row0 + r;
      const int k = k0 + 2 * jp;
      float v0 = 0.0f, v1 = 0.0f;
      if (e < M) {
        if (MODE == 0) {
          if (k < K)     v0 = A[(long)e * ldA + k];
          if (k + 1 < K) v1 = A[(long)e * ldA + k + 1];
        } else {
          const int d = dstI[e], s = srcI[e];
          const long db = (long)d * ldA, sb = (long)s * ldA;
          if (k < K)
            v0 = (k < F) ? A[db + k] : (A[sb + k - F] - A[db + k - F]);
          if (k + 1 < K)
            v1 = (k + 1 < F) ? A[db + k + 1] : (A[sb + k + 1 - F] - A[db + k + 1 - F]);
        }
      }
      As[r][jp] = pack2bf(v0, v1);
    }
    // -- stage W chunk, n-major (NL cols x 16 packed pairs) --
    for (int i = tid; i < NL * 16; i += 128) {
      const int n = i >> 4, jp = i & 15;
      const int k = k0 + 2 * jp;
      float v0 = 0.0f, v1 = 0.0f;
      if (n < Nout) {
        if (k < K)     v0 = W[(long)k * Nout + n];
        if (k + 1 < K) v1 = W[(long)(k + 1) * Nout + n];
      }
      Bs[n][jp] = pack2bf(v0, v1);
    }
    __syncthreads();

    // -- fragments: VGPR v <- packed index (v<4?v:v+4)+4*half --
    union Frag { v16bf v; unsigned u[8]; };
    Frag af;
#pragma unroll
    for (int v = 0; v < 8; ++v) {
      const int j = ((v < 4) ? v : v + 4) + 4 * half;
      af.u[v] = As[wrow + mloc][j];
    }
#pragma unroll
    for (int t = 0; t < NCT; ++t) {
      Frag bf;
#pragma unroll
      for (int v = 0; v < 8; ++v) {
        const int j = ((v < 4) ? v : v + 4) + 4 * half;
        bf.u[v] = Bs[t * 16 + mloc][j];
      }
      acc[t] = __builtin_amdgcn_wmma_f32_16x16x32_bf16(
          false, af.v, false, bf.v, (short)0, acc[t], false, false);
    }
    __syncthreads();
  }

  // -- store C (v8f layout: row = v + 8*half, col = lane&15) + bias --
#pragma unroll
  for (int t = 0; t < NCT; ++t) {
    const int n = t * 16 + mloc;
#pragma unroll
    for (int v = 0; v < 8; ++v) {
      const int r = row0 + wrow + v + 8 * half;
      if (r < M && n < Nout)
        out[(long)r * ldOut + colOff + n] = acc[t][v] + bias[n];
    }
  }
}

// ---------------------------------------------------------------------------
// Utility / elementwise kernels
// ---------------------------------------------------------------------------
__global__ void fill_f32_kernel(float* p, float v, int n) {
  int i = blockIdx.x * blockDim.x + threadIdx.x;
  if (i < n) p[i] = v;
}
__global__ void fill_u32_kernel(unsigned* p, unsigned v, int n) {
  int i = blockIdx.x * blockDim.x + threadIdx.x;
  if (i < n) p[i] = v;
}
__global__ void count_edges_kernel(const int* dst, float* cnt) {
  int e = blockIdx.x * blockDim.x + threadIdx.x;
  if (e < N_EDGES) atomicAdd(&cnt[dst[e]], 1.0f);
}

// per-channel sum / sumsq over M rows (C <= 96)
__global__ void col_stats_kernel(const float* __restrict__ x, int M, int C, int ld,
                                 float* __restrict__ sums) {
  __shared__ float ss[96], sq[96];
  const int tid = threadIdx.x;
  for (int j = tid; j < C; j += blockDim.x) { ss[j] = 0.f; sq[j] = 0.f; }
  __syncthreads();
  const int r = blockIdx.x * blockDim.x + tid;
  if (r < M) {
    for (int c = 0; c < C; ++c) {
      float v = x[(long)r * ld + c];
      atomicAdd(&ss[c], v);
      atomicAdd(&sq[c], v * v);
    }
  }
  __syncthreads();
  for (int j = tid; j < C; j += blockDim.x) {
    atomicAdd(&sums[j], ss[j]);
    atomicAdd(&sums[C + j], sq[j]);
  }
}
__global__ void stats_finalize_kernel(const float* sums, int C, float invM, float* minv) {
  int c = threadIdx.x;
  if (c >= C) return;
  float mean = sums[c] * invM;
  float var  = sums[C + c] * invM - mean * mean;
  minv[c]     = mean;
  minv[C + c] = rsqrtf(var + EPS_BN);
}

// EdgeConv: y = relu(bn(msg)) scattered (sum) into agg[dst]
__global__ void edge_bn_relu_scatter_kernel(const float* __restrict__ msg, int C,
                                            const float* minv, const float* g,
                                            const float* be, const int* dst,
                                            float* __restrict__ agg) {
  int e = blockIdx.x * blockDim.x + threadIdx.x;
  if (e >= N_EDGES) return;
  int d = dst[e];
  for (int c = 0; c < C; ++c) {
    float y = g[c] * (msg[(long)e * C + c] - minv[c]) * minv[C + c] + be[c];
    y = fmaxf(y, 0.0f);
    atomicAdd(&agg[(long)d * C + c], y);
  }
}
__global__ void edge_mean_write_kernel(const float* agg, const float* cnt, int C,
                                       float* out, int ldOut, int colOff) {
  int n = blockIdx.x * blockDim.x + threadIdx.x;
  if (n >= N_NODES) return;
  float inv = 1.0f / fmaxf(cnt[n], 1.0f);
  for (int c = 0; c < C; ++c)
    out[(long)n * ldOut + colOff + c] = agg[(long)n * C + c] * inv;
}

// GAT attention logits + running segment-max (ordered-uint atomicMax trick)
__global__ void gat_logits_kernel(const float* __restrict__ xl, const float* __restrict__ xr,
                                  const int* src, const int* dst, int O,
                                  const float* att, float* __restrict__ elog,
                                  unsigned* __restrict__ amaxU) {
  int idx = blockIdx.x * blockDim.x + threadIdx.x;
  if (idx >= N_SL) return;
  int s = (idx < N_EDGES) ? src[idx] : idx - N_EDGES;
  int d = (idx < N_EDGES) ? dst[idx] : idx - N_EDGES;
  for (int h = 0; h < HEADS; ++h) {
    float a = 0.0f;
    for (int c = 0; c < O; ++c) {
      float e = xl[(long)s * HEADS * O + h * O + c] + xr[(long)d * HEADS * O + h * O + c];
      e = (e > 0.0f) ? e : NEG_SLOPE * e;
      a += e * att[h * O + c];
    }
    elog[(long)idx * HEADS + h] = a;
    unsigned bits = __builtin_bit_cast(unsigned, a);
    unsigned key  = (bits & 0x80000000u) ? ~bits : (bits | 0x80000000u);
    atomicMax(&amaxU[d * HEADS + h], key);
  }
}
__global__ void gat_expsum_kernel(float* __restrict__ elog, const int* dst,
                                  const unsigned* amaxU, float* __restrict__ denom) {
  int idx = blockIdx.x * blockDim.x + threadIdx.x;
  if (idx >= N_SL) return;
  int d = (idx < N_EDGES) ? dst[idx] : idx - N_EDGES;
  for (int h = 0; h < HEADS; ++h) {
    unsigned key  = amaxU[d * HEADS + h];
    unsigned bits = (key & 0x80000000u) ? (key & 0x7FFFFFFFu) : ~key;
    float am = __builtin_bit_cast(float, bits);
    float ex = __expf(elog[(long)idx * HEADS + h] - am);
    elog[(long)idx * HEADS + h] = ex;
    atomicAdd(&denom[d * HEADS + h], ex);
  }
}
__global__ void gat_scatter_kernel(const float* __restrict__ elog,
                                   const float* __restrict__ xl,
                                   const int* src, const int* dst, int O,
                                   const float* denom, float* __restrict__ gacc) {
  int idx = blockIdx.x * blockDim.x + threadIdx.x;
  if (idx >= N_SL) return;
  int s = (idx < N_EDGES) ? src[idx] : idx - N_EDGES;
  int d = (idx < N_EDGES) ? dst[idx] : idx - N_EDGES;
  for (int h = 0; h < HEADS; ++h) {
    float alpha = elog[(long)idx * HEADS + h] / denom[d * HEADS + h];
    for (int c = 0; c < O; ++c)
      atomicAdd(&gacc[(long)d * HEADS * O + h * O + c],
                xl[(long)s * HEADS * O + h * O + c] * alpha);
  }
}
__global__ void gat_finalize_kernel(const float* gacc, int O, int concat,
                                    const float* bias, float* gpre) {
  int n = blockIdx.x * blockDim.x + threadIdx.x;
  if (n >= N_NODES) return;
  if (concat) {
    for (int j = 0; j < HEADS * O; ++j)
      gpre[(long)n * HEADS * O + j] = gacc[(long)n * HEADS * O + j] + bias[j];
  } else {
    for (int c = 0; c < O; ++c) {
      float m = (gacc[(long)n * HEADS * O + c] +
                 gacc[(long)n * HEADS * O + O + c] +
                 gacc[(long)n * HEADS * O + 2 * O + c]) * (1.0f / 3.0f);
      gpre[(long)n * O + c] = m + bias[c];
    }
  }
}
__global__ void bn_relu_write_kernel(const float* xin, int C, const float* minv,
                                     const float* g, const float* be,
                                     float* out, int ldOut, int colOff) {
  int n = blockIdx.x * blockDim.x + threadIdx.x;
  if (n >= N_NODES) return;
  for (int c = 0; c < C; ++c) {
    float y = g[c] * (xin[(long)n * C + c] - minv[c]) * minv[C + c] + be[c];
    out[(long)n * ldOut + colOff + c] = fmaxf(y, 0.0f);
  }
}

// Pack the 6 amortization-head weight matrices into one 32x64 GEMM operand.
__global__ void pack_heads_kernel(const float* muw, const float* mub,
                                  const float* vw,  const float* vb,
                                  const float* dw,  const float* db,
                                  const float* d1w, const float* d1b,
                                  const float* d2w, const float* d2b,
                                  const float* bw,  const float* bb,
                                  float* packW, float* packB) {
  int j = threadIdx.x;
  if (j >= 64) return;
  const float *w, *b; int off, wd;
  if (j < 2)       { w = muw; b = mub; off = j;      wd = 2;  }
  else if (j < 4)  { w = vw;  b = vb;  off = j - 2;  wd = 2;  }
  else if (j < 28) { w = dw;  b = db;  off = j - 4;  wd = 24; }
  else if (j < 40) { w = d1w; b = d1b; off = j - 28; wd = 12; }
  else if (j < 52) { w = d2w; b = d2b; off = j - 40; wd = 12; }
  else             { w = bw;  b = bb;  off = j - 52; wd = 12; }
  packB[j] = b[off];
  for (int k = 0; k < 32; ++k) packW[k * 64 + j] = w[k * wd + off];
}

// Deterministic hash Box-Muller (stand-in for jax.random.normal(key 42)).
static __device__ __forceinline__ unsigned pcg(unsigned x) {
  x ^= x >> 16; x *= 0x7feb352du; x ^= x >> 15; x *= 0x846ca68bu; x ^= x >> 16;
  return x;
}
static __device__ __forceinline__ float hash_normal(unsigned n, unsigned d) {
  unsigned a = pcg(n * 2u + d + 0x9E3779B9u);
  unsigned b = pcg(a ^ 0x85ebca6bu);
  float u1 = fmaxf((a >> 8) * (1.0f / 16777216.0f), 1e-7f);
  float u2 = (b >> 8) * (1.0f / 16777216.0f);
  return sqrtf(-2.0f * logf(u1)) * cosf(6.28318530718f * u2);
}

// Sylvester flows (HID=2, 6 flows), per node. ho row layout (64 cols):
// [0:2)=mu [2:4)=logvar [4:28)=full_d(i*12+j*6+k) [28:40)=diag1 [40:52)=diag2 [52:64)=b
__global__ void flows_kernel(const float* __restrict__ ho,
                             float* muO, float* lvO, float* ldjO,
                             float* z0O, float* zkO, float* zkbuf) {
  int n = blockIdx.x * blockDim.x + threadIdx.x;
  if (n >= N_NODES) return;
  const float* h = ho + (long)n * 64;
  float mu0 = h[0], mu1 = h[1], lv0 = h[2], lv1 = h[3];
  muO[n * 2] = mu0; muO[n * 2 + 1] = mu1;
  lvO[n * 2] = lv0; lvO[n * 2 + 1] = lv1;
  float z0 = mu0 + hash_normal(n, 0) * __expf(0.5f * lv0);
  float z1 = mu1 + hash_normal(n, 1) * __expf(0.5f * lv1);
  z0O[n * 2] = z0; z0O[n * 2 + 1] = z1;
  float za = z0, zb = z1, ldj = 0.0f;
#pragma unroll
  for (int k = 0; k < NFLOWS; ++k) {
    float r1_00 = tanhf(h[28 + k]);
    float r1_11 = tanhf(h[28 + 6 + k]);
    float r1_01 = h[4 + 0 * 12 + 1 * 6 + k];   // full_d[0,1,k]
    float r2_00 = tanhf(h[40 + k]);
    float r2_11 = tanhf(h[40 + 6 + k]);
    float r2_01 = h[4 + 1 * 12 + 0 * 6 + k];   // full_d[1,0,k] (swapaxes)
    float b0 = h[52 + k], b1 = h[52 + 6 + k];
    float zp0 = (k & 1) ? zb : za;
    float zp1 = (k & 1) ? za : zb;
    float q0 = r2_00 * zp0 + r2_01 * zp1 + b0;
    float q1 = r2_11 * zp1 + b1;
    float h0 = tanhf(q0), h1 = tanhf(q1);
    float zn0 = h0 * r1_00 + h1 * r1_01;
    float zn1 = h1 * r1_11;
    if (k & 1) { float t = zn0; zn0 = zn1; zn1 = t; }
    zn0 += za; zn1 += zb;
    float dj0 = (1.0f - h0 * h0) * (r1_00 * r2_00) + 1.0f;
    float dj1 = (1.0f - h1 * h1) * (r1_11 * r2_11) + 1.0f;
    ldj += logf(fabsf(dj0)) + logf(fabsf(dj1));
    za = zn0; zb = zn1;
  }
  ldjO[n] = ldj;
  zkO[n * 2] = za; zkO[n * 2 + 1] = zb;
  zkbuf[n * 2] = za; zkbuf[n * 2 + 1] = zb;
}

// xdec = lin_out_2(relu(lin_out(hdec)))
__global__ void out_head_kernel(const float* hdec, const float* w1, const float* b1,
                                const float* w2, const float* b2, float* xdec) {
  int n = blockIdx.x * blockDim.x + threadIdx.x;
  if (n >= N_NODES) return;
  float hv[8], y[4];
  for (int i = 0; i < 8; ++i) hv[i] = hdec[(long)n * 8 + i];
  for (int o = 0; o < 4; ++o) {
    float a = b1[o];
    for (int i = 0; i < 8; ++i) a += hv[i] * w1[i * 4 + o];
    y[o] = fmaxf(a, 0.0f);
  }
  for (int o = 0; o < 4; ++o) {
    float a = b2[o];
    for (int i = 0; i < 4; ++i) a += y[i] * w2[i * 4 + o];
    xdec[(long)n * 4 + o] = a;
  }
}

// ---------------------------------------------------------------------------
// Host-side orchestration
// ---------------------------------------------------------------------------
struct Ctx {
  hipStream_t st;
  const int *src, *dst;
  float *nodeA, *nodeB, *msg, *agg, *cnt, *xl, *xr, *elog, *denom, *gacc, *gpre;
  float *sums, *minv, *packW, *packB, *headout, *zkbuf;
  unsigned* amaxU;
};
static inline int cdiv(int a, int b) { return (a + b - 1) / b; }

static void fillF(hipStream_t st, float* p, float v, int n) {
  fill_f32_kernel<<<cdiv(n, 256), 256, 0, st>>>(p, v, n);
}
static void fillU(hipStream_t st, unsigned* p, unsigned v, int n) {
  fill_u32_kernel<<<cdiv(n, 256), 256, 0, st>>>(p, v, n);
}

static void gemm(const Ctx& c, int mode, int M, int K, int F, int Nout,
                 const float* A, int ldA, const float* W, const float* b,
                 float* out, int ldOut, int colOff) {
  int nct = (Nout + 15) / 16;
  dim3 g(cdiv(M, 64)), blk(128);
#define GL(NCT, MODE) gemm_bf16_wmma_kernel<NCT, MODE><<<g, blk, 0, c.st>>>( \
      M, K, F, Nout, A, ldA, c.src, c.dst, W, b, out, ldOut, colOff)
  if (mode == 0) {
    switch (nct) { case 1: GL(1,0); break; case 2: GL(2,0); break;
                   case 3: GL(3,0); break; case 4: GL(4,0); break; default: GL(6,0); }
  } else {
    switch (nct) { case 1: GL(1,1); break; case 2: GL(2,1); break;
                   case 3: GL(3,1); break; case 4: GL(4,1); break; default: GL(6,1); }
  }
#undef GL
}

// EdgeConv: msg GEMM -> BN over E rows -> relu -> segment-mean into out cols
static void run_edge(const Ctx& c, const float* in, int ldIn, int F,
                     const float* w, const float* b, const float* g, const float* be,
                     int Nout, float* out, int ldOut, int colOff) {
  gemm(c, 1, N_EDGES, 2 * F, F, Nout, in, ldIn, w, b, c.msg, Nout, 0);
  fillF(c.st, c.sums, 0.f, 2 * Nout);
  col_stats_kernel<<<cdiv(N_EDGES, 256), 256, 0, c.st>>>(c.msg, N_EDGES, Nout, Nout, c.sums);
  stats_finalize_kernel<<<1, 128, 0, c.st>>>(c.sums, Nout, 1.0f / N_EDGES, c.minv);
  fillF(c.st, c.agg, 0.f, N_NODES * Nout);
  edge_bn_relu_scatter_kernel<<<cdiv(N_EDGES, 256), 256, 0, c.st>>>(
      c.msg, Nout, c.minv, g, be, c.dst, c.agg);
  edge_mean_write_kernel<<<cdiv(N_NODES, 256), 256, 0, c.st>>>(
      c.agg, c.cnt, Nout, out, ldOut, colOff);
}

// GATConv: two linears -> softmax attention over (edges + self loops) ->
// weighted segment-sum -> (concat|head-mean)+bias -> BN -> relu -> out cols
static void run_gat(const Ctx& c, const float* in, int ldIn, int F, int O, int concat,
                    const float* lw, const float* lb, const float* rw, const float* rb,
                    const float* att, const float* bias, const float* g, const float* be,
                    float* out, int ldOut, int colOff) {
  const int HO = HEADS * O;
  gemm(c, 0, N_NODES, F, F, HO, in, ldIn, lw, lb, c.xl, HO, 0);
  gemm(c, 0, N_NODES, F, F, HO, in, ldIn, rw, rb, c.xr, HO, 0);
  fillU(c.st, c.amaxU, 0u, N_NODES * HEADS);
  fillF(c.st, c.denom, 0.f, N_NODES * HEADS);
  gat_logits_kernel<<<cdiv(N_SL, 256), 256, 0, c.st>>>(c.xl, c.xr, c.src, c.dst, O,
                                                       att, c.elog, c.amaxU);
  gat_expsum_kernel<<<cdiv(N_SL, 256), 256, 0, c.st>>>(c.elog, c.dst, c.amaxU, c.denom);
  fillF(c.st, c.gacc, 0.f, N_NODES * HO);
  gat_scatter_kernel<<<cdiv(N_SL, 256), 256, 0, c.st>>>(c.elog, c.xl, c.src, c.dst, O,
                                                        c.denom, c.gacc);
  gat_finalize_kernel<<<cdiv(N_NODES, 256), 256, 0, c.st>>>(c.gacc, O, concat, bias, c.gpre);
  const int Dg = concat ? HO : O;
  fillF(c.st, c.sums, 0.f, 2 * Dg);
  col_stats_kernel<<<cdiv(N_NODES, 256), 256, 0, c.st>>>(c.gpre, N_NODES, Dg, Dg, c.sums);
  stats_finalize_kernel<<<1, 128, 0, c.st>>>(c.sums, Dg, 1.0f / N_NODES, c.minv);
  bn_relu_write_kernel<<<cdiv(N_NODES, 256), 256, 0, c.st>>>(c.gpre, Dg, c.minv, g, be,
                                                             out, ldOut, colOff);
}

extern "C" void kernel_launch(void* const* d_in, const int* in_sizes, int n_in,
                              void* d_out, int out_size, void* d_ws, size_t ws_size,
                              hipStream_t stream) {
  (void)in_sizes; (void)n_in; (void)out_size; (void)ws_size;
  // d_in[0]=x (N,4) f32; d_in[1]=edge_index (2,E) i32; params flattened as a
  // JAX pytree (dict keys sorted alphabetically at every level, {b,w}/{be,g}
  // within leaves). Leaf index map (hardcoded below, starting at 2).
  auto P = [&](int i) { return (const float*)d_in[i]; };
  const float* x = (const float*)d_in[0];
  const int* ei  = (const int*)d_in[1];

  // workspace arena
  float* w = (float*)d_ws;
  size_t o = 0;
  float* nodeA = w + o; o += (size_t)N_NODES * 128;
  float* nodeB = w + o; o += (size_t)N_NODES * 128;
  float* msg   = w + o; o += (size_t)N_EDGES * 32;
  float* agg   = w + o; o += (size_t)N_NODES * 32;
  float* cnt   = w + o; o += (size_t)N_NODES;
  float* xl    = w + o; o += (size_t)N_NODES * 96;
  float* xr    = w + o; o += (size_t)N_NODES * 96;
  float* elog  = w + o; o += (size_t)N_SL * 3;
  unsigned* amaxU = (unsigned*)(w + o); o += (size_t)N_NODES * 3;
  float* denom = w + o; o += (size_t)N_NODES * 3;
  float* gacc  = w + o; o += (size_t)N_NODES * 96;
  float* gpre  = w + o; o += (size_t)N_NODES * 96;
  float* sums  = w + o; o += 256;
  float* minv  = w + o; o += 256;
  float* packW = w + o; o += 32 * 64;
  float* packB = w + o; o += 64;
  float* headout = w + o; o += (size_t)N_NODES * 64;
  float* zkbuf = w + o; o += (size_t)N_NODES * 2;

  Ctx c{stream, ei, ei + N_EDGES, nodeA, nodeB, msg, agg, cnt, xl, xr, elog,
        denom, gacc, gpre, sums, minv, packW, packB, headout, zkbuf, amaxU};

  // in-degree counts (shared by every EdgeConv layer)
  fillF(stream, cnt, 0.f, N_NODES);
  count_edges_kernel<<<cdiv(N_EDGES, 256), 256, 0, stream>>>(c.dst, c.cnt);

  // ---------------- encoder inception ----------------
  // L0: x (N,4) -> nodeA (N,128)  [xe cols 0:32 | xg cols 32:128]
  run_edge(c, x, 4, 4,  P(49), P(48), P(47), P(46), 32, nodeA, 128, 0);
  run_gat (c, x, 4, 4, 32, 1, P(63), P(62), P(65), P(64), P(60), P(61), P(59), P(58),
           nodeA, 128, 32);
  // L1: nodeA -> nodeB
  run_edge(c, nodeA, 128, 128, P(53), P(52), P(51), P(50), 32, nodeB, 128, 0);
  run_gat (c, nodeA, 128, 128, 32, 1, P(71), P(70), P(73), P(72), P(68), P(69), P(67), P(66),
           nodeB, 128, 32);
  // L2: nodeB -> henc = nodeA (N,32) [xe 0:16 | xg 16:32]
  run_edge(c, nodeB, 128, 128, P(57), P(56), P(55), P(54), 16, nodeA, 32, 0);
  run_gat (c, nodeB, 128, 128, 16, 0, P(79), P(78), P(81), P(80), P(76), P(77), P(75), P(74),
           nodeA, 32, 16);

  // ---------------- amortization heads + flows ----------------
  pack_heads_kernel<<<1, 64, 0, stream>>>(P(87), P(86), P(89), P(88), P(5), P(4),
                                          P(7), P(6), P(9), P(8), P(3), P(2),
                                          packW, packB);
  gemm(c, 0, N_NODES, 32, 32, 64, nodeA, 32, packW, packB, headout, 64, 0);

  float* outF = (float*)d_out;
  float* xdecO = outF;                 // 20000*4
  float* muO   = outF + 80000;         // 20000*2
  float* lvO   = outF + 120000;        // 20000*2
  float* ldjO  = outF + 160000;        // 20000
  float* z0O   = outF + 180000;        // 20000*2
  float* zkO   = outF + 220000;        // 20000*2
  flows_kernel<<<cdiv(N_NODES, 256), 256, 0, stream>>>(headout, muO, lvO, ldjO,
                                                       z0O, zkO, zkbuf);

  // ---------------- decoder inception ----------------
  // L0: zk (N,2) -> nodeB (N,128)
  run_edge(c, zkbuf, 2, 2,  P(13), P(12), P(11), P(10), 32, nodeB, 128, 0);
  run_gat (c, zkbuf, 2, 2, 32, 1, P(27), P(26), P(29), P(28), P(24), P(25), P(23), P(22),
           nodeB, 128, 32);
  // L1: nodeB -> nodeA
  run_edge(c, nodeB, 128, 128, P(17), P(16), P(15), P(14), 32, nodeA, 128, 0);
  run_gat (c, nodeB, 128, 128, 32, 1, P(35), P(34), P(37), P(36), P(32), P(33), P(31), P(30),
           nodeA, 128, 32);
  // L2: nodeA -> hdec = nodeB (N,8) [xe 0:4 | xg 4:8]
  run_edge(c, nodeA, 128, 128, P(21), P(20), P(19), P(18), 4, nodeB, 8, 0);
  run_gat (c, nodeA, 128, 128, 4, 0, P(43), P(42), P(45), P(44), P(40), P(41), P(39), P(38),
           nodeB, 8, 4);

  // ---------------- output head ----------------
  out_head_kernel<<<cdiv(N_NODES, 256), 256, 0, stream>>>(nodeB, P(83), P(82),
                                                          P(85), P(84), xdecO);
}
